// RayObsGraph_30451318128961
// MI455X (gfx1250) — matrix operand order repaired and checked by reading, not compiled, not run on hardware
//
#include <hip/hip_runtime.h>
#include <hip/hip_fp16.h>

typedef __attribute__((ext_vector_type(16))) _Float16 v16h;
typedef __attribute__((ext_vector_type(8)))  float    v8f;
typedef __attribute__((ext_vector_type(4)))  _Float16 v4h;

#define B_    512
#define G_    100
#define D_    512
#define H_    256
#define O_    1024
#define NOUT_ 18

// ---------------------------------------------------------------------------
// prep_x: Xh[b,g,d] = f16( g == n[b] ? flat[b,d] : nodes[b,g,d] )
// ---------------------------------------------------------------------------
__global__ void prep_x_kernel(const float* __restrict__ flat,
                              const int*   __restrict__ num_nodes,
                              const float* __restrict__ nodes,
                              _Float16*    __restrict__ Xh) {
  int t    = blockIdx.x * blockDim.x + threadIdx.x;   // one thread per 4 elems
  int idx4 = t * 4;
  if (idx4 >= B_ * G_ * D_) return;
  int b   = idx4 / (G_ * D_);
  int rem = idx4 - b * (G_ * D_);
  int g   = rem / D_;
  int d   = rem - g * D_;
  int n   = num_nodes[b];
  const float* src = (g == n) ? (flat + (size_t)b * D_ + d) : (nodes + idx4);
  float4 f = *(const float4*)src;
  v4h h;
  h.x = (_Float16)f.x; h.y = (_Float16)f.y;
  h.z = (_Float16)f.z; h.w = (_Float16)f.w;
  *(v4h*)(Xh + idx4) = h;
}

// ---------------------------------------------------------------------------
// Weight repack (N-major, K contiguous) so GEMM B-tiles load coalesced.
// W0t[n][k] : n<256 -> w0_rel[k][n], else w0_root[k][n-256]   (N=512, K=512)
// W1t[n][k] : k<256 -> w1_rel[k][n], else w1_root[k-256][n]   (N=1024, K=512)
// ---------------------------------------------------------------------------
__global__ void prep_w0_kernel(const float* __restrict__ wrel,
                               const float* __restrict__ wroot,
                               _Float16*    __restrict__ Wt) {
  int i = blockIdx.x * blockDim.x + threadIdx.x;
  if (i >= 512 * 512) return;
  int n = i >> 9, k = i & 511;
  float v = (n < 256) ? wrel[(size_t)k * H_ + n] : wroot[(size_t)k * H_ + (n - 256)];
  Wt[i] = (_Float16)v;
}

__global__ void prep_w1_kernel(const float* __restrict__ wrel,
                               const float* __restrict__ wroot,
                               _Float16*    __restrict__ Wt) {
  int i = blockIdx.x * blockDim.x + threadIdx.x;
  if (i >= 1024 * 512) return;
  int n = i >> 9, k = i & 511;
  float v = (k < 256) ? wrel[(size_t)k * O_ + n] : wroot[(size_t)(k - 256) * O_ + n];
  Wt[i] = (_Float16)v;
}

// ---------------------------------------------------------------------------
// f16 WMMA GEMM:  C[M,N] = A[M,K] @ Bt[N,K]^T
// 128x128 block tile, BK=32, 8 waves (wave32), 8 v_wmma_f32_16x16x32_f16/wave/step.
// Double-buffered LDS staging via CDNA5 async global->LDS copies:
//   issue tile k+1, s_wait_asynccnt<=4 (async completes in-order, 4/thread/tile)
//   => tile k's copies are done while k+1 streams behind the WMMAs.
// Bias/ReLU/output-precision are compile-time template flags (clean epilogue).
// ---------------------------------------------------------------------------
#define BM 128
#define BN 128
#define BK 32
#define LDSK 40   // 32 + 8 halves pad: 80B row stride, b128-aligned, conflict-free
#define TILE_HALVES (BM * LDSK)

union Frag { v16h h; uint4 q[2]; };

template<bool HAS_BIAS, bool DO_RELU, bool OUT_F16>
__global__ void __launch_bounds__(256)
gemm_f16_kernel(const _Float16* __restrict__ A,
                const _Float16* __restrict__ Bt,     // [N][K]
                void*           __restrict__ Cout,
                int M, int N, int K,
                const float* __restrict__ bias) {
  __shared__ _Float16 alds[2 * TILE_HALVES];
  __shared__ _Float16 blds[2 * TILE_HALVES];

  const int tid   = threadIdx.x;
  const int lane  = tid & 31;
  const int wid   = tid >> 5;
  const int wm    = wid & 3;        // 4 wave-rows  x 32 rows
  const int wn    = wid >> 2;       // 2 wave-cols  x 64 cols
  const int m0    = blockIdx.y * BM;
  const int n0    = blockIdx.x * BN;
  const int lhalf = lane >> 4;      // 0 | 1
  const int lmod  = lane & 15;

  // Async-stage one 128x32 A tile + B tile into LDS buffer `buf`.
  // LDS byte address = low 32 bits of the flat shared pointer (aperture rule).
  auto stage = [&](int kt, int buf) {
    const _Float16* al = alds + buf * TILE_HALVES;
    const _Float16* bl = blds + buf * TILE_HALVES;
    int k0 = kt * BK;
#pragma unroll
    for (int s = 0; s < 2; s++) {
      int chunk = tid + s * 256;
      int row   = chunk >> 2;
      int kc    = (chunk & 3) << 3;       // 0,8,16,24 halves (16B chunks)
      unsigned lA = (unsigned)(size_t)(al + row * LDSK + kc);
      unsigned long long gA =
          (unsigned long long)(size_t)(A + (size_t)(m0 + row) * K + k0 + kc);
      asm volatile("global_load_async_to_lds_b128 %0, %1, off"
                   :: "v"(lA), "v"(gA) : "memory");
      unsigned lB = (unsigned)(size_t)(bl + row * LDSK + kc);
      unsigned long long gB =
          (unsigned long long)(size_t)(Bt + (size_t)(n0 + row) * K + k0 + kc);
      asm volatile("global_load_async_to_lds_b128 %0, %1, off"
                   :: "v"(lB), "v"(gB) : "memory");
    }
  };

  v8f acc[2][4];
  for (int i = 0; i < 2; i++)
    for (int j = 0; j < 4; j++)
      for (int r = 0; r < 8; r++) acc[i][j][r] = 0.0f;

  const int KT = K / BK;
  stage(0, 0);                         // prologue: tile 0 in flight

  for (int kt = 0; kt < KT; kt++) {
    int buf = kt & 1;
    if (kt + 1 < KT) {
      stage(kt + 1, buf ^ 1);          // stream next tile behind the WMMAs
      asm volatile("s_wait_asynccnt 0x4" ::: "memory");   // tile kt landed
    } else {
      asm volatile("s_wait_asynccnt 0x0" ::: "memory");
    }
    __syncthreads();                   // tile kt visible to all waves

    const _Float16* al = alds + buf * TILE_HALVES;
    const _Float16* bl = blds + buf * TILE_HALVES;

    Frag af[2], bf[4];
    // A fragment: lane<16 -> K {0..7,16..23}; lane>=16 -> +8  (ISA 16-bit A layout)
#pragma unroll
    for (int tm = 0; tm < 2; tm++) {
      const _Float16* p = al + (wm * 32 + tm * 16 + lmod) * LDSK + lhalf * 8;
      af[tm].q[0] = *(const uint4*)p;
      af[tm].q[1] = *(const uint4*)(p + 16);
    }
    // B fragment: lane gives N; lane<16 -> K {0..7,16..23}; lane>=16 -> +8
#pragma unroll
    for (int tn = 0; tn < 4; tn++) {
      const _Float16* p = bl + (wn * 64 + tn * 16 + lmod) * LDSK + lhalf * 8;
      bf[tn].q[0] = *(const uint4*)p;
      bf[tn].q[1] = *(const uint4*)(p + 16);
    }
#pragma unroll
    for (int tm = 0; tm < 2; tm++)
#pragma unroll
      for (int tn = 0; tn < 4; tn++)
        acc[tm][tn] = __builtin_amdgcn_wmma_f32_16x16x32_f16(
            false, af[tm].h, false, bf[tn].h, (short)0, acc[tm][tn], false, false);

    __syncthreads();                   // all reads of buf done before reuse
  }

  // epilogue: 32-bit C/D layout -> row = r + 8*lhalf, col = lmod
#pragma unroll
  for (int tm = 0; tm < 2; tm++) {
#pragma unroll
    for (int tn = 0; tn < 4; tn++) {
      int col   = n0 + wn * 64 + tn * 16 + lmod;
      int row0  = m0 + wm * 32 + tm * 16 + lhalf * 8;
      float bv  = HAS_BIAS ? bias[col] : 0.0f;
      size_t ix = (size_t)row0 * N + col;
#pragma unroll
      for (int r = 0; r < 8; r++) {
        float v = acc[tm][tn][r] + bv;
        if (DO_RELU) v = fmaxf(v, 0.0f);
        if (OUT_F16) ((_Float16*)Cout)[ix] = (_Float16)v;
        else         ((float*)Cout)[ix] = v;
        ix += N;
      }
    }
  }
}

// ---------------------------------------------------------------------------
// h[b,i,c] = relu( sum_j A'[b,j,i]*U[b,j,c] + V[b,i,c] + b0[c] )
// UV (f16) cols 0..255 = U (X@w0_rel), 256..511 = V (X@w0_root).
// ---------------------------------------------------------------------------
__global__ void graph_agg0_kernel(const _Float16* __restrict__ UV,
                                  const int*      __restrict__ adj,
                                  const int*      __restrict__ num_nodes,
                                  const float*    __restrict__ b0,
                                  _Float16*       __restrict__ hbuf) {
  int i = blockIdx.x;          // node 0..99
  int b = blockIdx.y;
  int c = threadIdx.x;         // 0..255
  int n = num_nodes[b];
  int cap = n > 1;
  int p = (n > 0) ? n - 1 : 0;
  const _Float16* uvb  = UV  + (size_t)b * G_ * 512;
  const int*      adjb = adj + (size_t)b * G_ * G_;
  float acc = (float)uvb[(size_t)i * 512 + 256 + c] + b0[c];
  for (int j = 0; j < G_; j++) {
    int e = adjb[j * G_ + i];
    if (j == n && i == n) e = 1;
    if (cap && ((j == p && i == n) || (j == n && i == p))) e = 1;
    if (e) acc += (float)uvb[(size_t)j * 512 + c];   // wave-uniform branch
  }
  hbuf[((size_t)b * G_ + i) * H_ + c] = (_Float16)fmaxf(acc, 0.0f);
}

// ---------------------------------------------------------------------------
// A2[b] = [ aggn | hn ] : aggn = sum_j A'[b,j,n] h[b,j,:], hn = h[b,n,:]
// ---------------------------------------------------------------------------
__global__ void stageB_kernel(const _Float16* __restrict__ hbuf,
                              const int*      __restrict__ adj,
                              const int*      __restrict__ num_nodes,
                              _Float16*       __restrict__ A2) {
  int b = blockIdx.x;
  int c = threadIdx.x;         // 0..255
  int n = num_nodes[b];
  int cap = n > 1;
  int p = (n > 0) ? n - 1 : 0;
  const _Float16* hb   = hbuf + (size_t)b * G_ * H_;
  const int*      adjb = adj  + (size_t)b * G_ * G_;
  float acc = 0.0f;
  for (int j = 0; j < G_; j++) {
    int e = adjb[j * G_ + n];
    if (j == n) e = 1;
    if (cap && j == p) e = 1;
    if (e) acc += (float)hb[(size_t)j * H_ + c];
  }
  A2[(size_t)b * 512 + c]       = (_Float16)acc;
  A2[(size_t)b * 512 + 256 + c] = hb[(size_t)n * H_ + c];
}

// ---------------------------------------------------------------------------
// logits/values head: per batch, 19 length-1024 dot products (wave-per-output)
// ---------------------------------------------------------------------------
__global__ void head_kernel(const float* __restrict__ tgt,
                            const float* __restrict__ w_logit,
                            const float* __restrict__ b_logit,
                            const float* __restrict__ w_val,
                            const float* __restrict__ b_val,
                            float*       __restrict__ out) {
  int b    = blockIdx.x;
  int lane = threadIdx.x & 31;
  int wid  = threadIdx.x >> 5;           // 8 waves
  const float* t = tgt + (size_t)b * O_;
  for (int o = wid; o < NOUT_ + 1; o += 8) {
    float s = 0.0f;
    if (o < NOUT_) {
      for (int c = lane; c < O_; c += 32) s += t[c] * w_logit[(size_t)c * NOUT_ + o];
    } else {
      for (int c = lane; c < O_; c += 32) s += t[c] * w_val[c];
    }
    for (int off = 16; off > 0; off >>= 1) s += __shfl_down(s, off, 32);
    if (lane == 0) {
      if (o < NOUT_) out[(size_t)b * NOUT_ + o] = s + b_logit[o];
      else           out[(size_t)B_ * NOUT_ + b] = s + b_val[0];
    }
  }
}

// ---------------------------------------------------------------------------
extern "C" void kernel_launch(void* const* d_in, const int* in_sizes, int n_in,
                              void* d_out, int out_size, void* d_ws, size_t ws_size,
                              hipStream_t stream) {
  (void)in_sizes; (void)n_in; (void)out_size; (void)ws_size;

  const float* flat      = (const float*)d_in[0];
  const int*   num_nodes = (const int*)  d_in[1];
  const float* nodes     = (const float*)d_in[2];
  const int*   adj       = (const int*)  d_in[3];
  const float* w0_rel    = (const float*)d_in[4];
  const float* w0_root   = (const float*)d_in[5];
  const float* b0        = (const float*)d_in[6];
  const float* w1_rel    = (const float*)d_in[7];
  const float* w1_root   = (const float*)d_in[8];
  const float* b1        = (const float*)d_in[9];
  const float* w_logit   = (const float*)d_in[10];
  const float* b_logit   = (const float*)d_in[11];
  const float* w_val     = (const float*)d_in[12];
  const float* b_val     = (const float*)d_in[13];

  char* ws = (char*)d_ws;
  size_t off = 0;
  _Float16* Xh  = (_Float16*)(ws + off); off += (size_t)B_ * G_ * D_ * 2;   // 52.4 MB
  _Float16* W0t = (_Float16*)(ws + off); off += (size_t)512 * 512 * 2;      // 0.5 MB
  _Float16* W1t = (_Float16*)(ws + off); off += (size_t)1024 * 512 * 2;     // 1 MB
  _Float16* UV  = (_Float16*)(ws + off); off += (size_t)B_ * G_ * 512 * 2;  // 52.4 MB
  _Float16* hb  = (_Float16*)(ws + off); off += (size_t)B_ * G_ * H_ * 2;   // 26.2 MB
  _Float16* A2  = (_Float16*)(ws + off); off += (size_t)B_ * 512 * 2;       // 0.5 MB
  float*    tgt = (float*)   (ws + off); off += (size_t)B_ * O_ * 4;        // 2.1 MB

  // graph build folded into f16 conversion
  prep_x_kernel<<<dim3((B_ * G_ * D_ / 4 + 255) / 256), 256, 0, stream>>>(
      flat, num_nodes, nodes, Xh);
  prep_w0_kernel<<<(512 * 512 + 255) / 256, 256, 0, stream>>>(w0_rel, w0_root, W0t);
  prep_w1_kernel<<<(1024 * 512 + 255) / 256, 256, 0, stream>>>(w1_rel, w1_root, W1t);

  // GEMM0: UV[51200,512] = Xh @ [w0_rel|w0_root]   (grid 4 x 400), f16 out
  gemm_f16_kernel<false, false, true><<<dim3(512 / BN, (B_ * G_) / BM), 256, 0, stream>>>(
      Xh, W0t, (void*)UV, B_ * G_, 512, 512, nullptr);

  // h = relu(A'^T U + V + b0)
  graph_agg0_kernel<<<dim3(G_, B_), H_, 0, stream>>>(UV, adj, num_nodes, b0, hb);

  // target-node aggregation for layer 1
  stageB_kernel<<<B_, H_, 0, stream>>>(hb, adj, num_nodes, A2);

  // GEMM1: tgt[512,1024] = relu(A2 @ [[w1_rel];[w1_root]] + b1), f32 out
  gemm_f16_kernel<true, true, false><<<dim3(O_ / BN, B_ / BM), 256, 0, stream>>>(
      A2, W1t, (void*)tgt, B_, O_, 512, b1);

  // logits + values -> d_out (logits [512*18] then values [512])
  head_kernel<<<B_, 256, 0, stream>>>(tgt, w_logit, b_logit, w_val, b_val, (float*)d_out);
}
